// PostProcessSegm_5368709120351
// MI455X (gfx1250) — matrix operand order repaired and checked by reading, not compiled, not run on hardware
//
#include <hip/hip_runtime.h>
#include <hip/hip_bf16.h>
#include <cstdint>

// Problem constants (from setup_inputs)
#define NQ    300
#define NC    42
#define NL    (NQ*NC)    // 12600 logits per batch
#define TOPKN 100
#define MT    3
#define MH    72
#define MW    128
#define MWP   132        // padded LDS row stride (132*4B: 16B-aligned, 132%64=4 -> no bank conflict)
#define MHP   80         // padded rows of out_w (removes divergent store guard)
#define UH    272        // img_h (cropped rows of the x4 upsample)
#define UW    480        // img_w (cropped cols of the x4 upsample)
#define OH    544        // orig_h = 2*UH (nearest x2)
#define OW    960        // orig_w = 2*UW
#define OUT_MASK_OFF 1200  // 200 scores + 200 labels + 800 refs

typedef __attribute__((ext_vector_type(16))) _Float16 v16h;
typedef __attribute__((ext_vector_type(8)))  float    v8f;
typedef __attribute__((ext_vector_type(4)))  float    f32x4;
typedef __attribute__((ext_vector_type(4)))  unsigned int u32x4;
typedef __attribute__((ext_vector_type(8)))  int          i32x8;
typedef __attribute__((ext_vector_type(4)))  int          i32x4;

__device__ __forceinline__ int imin(int a, int b) { return a < b ? a : b; }
__device__ __forceinline__ int imax(int a, int b) { return a > b ? a : b; }

// ---------------------------------------------------------------------------
// Kernel 1: per-batch top-100 (on raw logits; sigmoid is monotone), emit
// scores / labels / gathered reference points, and topk_boxes to workspace.
// ---------------------------------------------------------------------------
__global__ void __launch_bounds__(256)
topk_kernel(const float* __restrict__ logits,
            const float* __restrict__ ref_pts,
            float* __restrict__ out,
            int* __restrict__ boxes)
{
    __shared__ float vals[NL];
    __shared__ float rv[256];
    __shared__ int   ri[256];
    const int b = blockIdx.x, tid = threadIdx.x;

    for (int i = tid; i < NL; i += 256) vals[i] = logits[b * NL + i];
    __syncthreads();

    for (int k = 0; k < TOPKN; ++k) {
        float best = -3.4e38f; int bi = 0x7fffffff;
        for (int i = tid; i < NL; i += 256) {       // ascending scan => lowest idx wins ties
            float v = vals[i];
            if (v > best) { best = v; bi = i; }
        }
        rv[tid] = best; ri[tid] = bi;
        __syncthreads();
        for (int s = 128; s > 0; s >>= 1) {
            if (tid < s) {
                float ov = rv[tid + s]; int oi = ri[tid + s];
                if (ov > rv[tid] || (ov == rv[tid] && oi < ri[tid])) { rv[tid] = ov; ri[tid] = oi; }
            }
            __syncthreads();
        }
        if (tid == 0) {
            int idx = ri[0];
            out[b * TOPKN + k]       = 1.0f / (1.0f + __expf(-rv[0]));   // score
            out[200 + b * TOPKN + k] = (float)(idx % NC);                // label
            int box = idx / NC;
            const float* rp = ref_pts + (b * NQ + box) * 4;
            float* ro = out + 400 + (b * TOPKN + k) * 4;                 // refs
            ro[0] = rp[0]; ro[1] = rp[1]; ro[2] = rp[2]; ro[3] = rp[3];
            boxes[b * TOPKN + k] = box;
            vals[idx] = -3.4e38f;
        }
        __syncthreads();
    }
}

// ---------------------------------------------------------------------------
// Kernel 2: one block per (b, q, t) slab.
//   Phase A: TDM tensor_load_to_lds of the gathered 72x128 f32 tile with
//            hardware row padding (128 DWORDs -> +4 DWORDs), TENSORcnt wait.
//            Fallback: per-wave global_load_async_to_lds_b128 (ASYNCcnt).
//   Phase B: width x4 bilinear (128 -> 480 cols) as v_wmma_f32_16x16x32_f16:
//            each 16x16 output tile needs a K-window of <=6 input cols, so it
//            is exactly one WMMA with analytic triangle-filter weights; the
//            weight fragment is built once per column tile and reused by the
//            5 row tiles (jc-outer / mt-inner).
//   Phase C: height x4 bilinear (2-tap lerp) + sign threshold + x2 nearest
//            replication, non-temporal b128 stores (1.25 GB write-once).
// ---------------------------------------------------------------------------
__global__ void __launch_bounds__(256)
masks_kernel(const float* __restrict__ pmasks,
             const int* __restrict__ boxes,
             float* __restrict__ out)
{
    extern __shared__ float smem[];
    float* in_tile = smem;               // MH*MWP = 9504 floats (padded rows)
    float* out_w   = smem + MH * MWP;    // MHP*UW = 38400 floats

    const int tid  = threadIdx.x;
    const int slab = blockIdx.x;                 // ((b*100 + q)*3 + t)
    const int b    = slab / (TOPKN * MT);
    const int rem  = slab % (TOPKN * MT);
    const int q    = rem / MT;
    const int t    = rem % MT;
    const int box  = boxes[b * TOPKN + q];
    const float* src = pmasks + ((((size_t)b * NQ + box) * MT + t) * (size_t)(MH * MW));

#if __has_builtin(__builtin_amdgcn_tensor_load_to_lds) && __has_builtin(__builtin_amdgcn_s_wait_tensorcnt)
    // ---- Phase A (TDM): one DMA descriptor, wave 0 only (TDM ignores EXEC) ----
    if (tid < 32) {
        const unsigned           lds_base = (unsigned)(uintptr_t)in_tile;
        const unsigned long long ga       = (unsigned long long)(uintptr_t)src;
        // D# group 0: count=1 | lds_addr | global_addr[56:0] | type=2
        u32x4 g0 = { 1u,
                     lds_base,
                     (unsigned)(ga & 0xffffffffull),
                     (unsigned)((ga >> 32) & 0x01ffffffull) | (2u << 30) };
        // D# group 1: data_size=4B(code 2) | pad_enable | pad_interval=128dw(code 6)
        //             | pad_amount=4dw(code 3) | tensor 128x72 | tile 128x72 | stride0=128
        i32x8 g1 = { (int)((2u << 16) | (1u << 20) | (6u << 22) | (3u << 25)),
                     (int)(128u << 16),   // tensor_dim0[15:0] in bits[31:16]
                     (int)(72u  << 16),   // tensor_dim0[31:16]=0 | tensor_dim1[15:0]
                     (int)(128u << 16),   // tensor_dim1[31:16]=0 | tile_dim0=128
                     (int)72,             // tile_dim1=72 | tile_dim2=0
                     (int)128,            // tensor_dim0_stride low 32
                     0, 0 };
        i32x4 gz4 = { 0, 0, 0, 0 };
        i32x8 gz8 = { 0, 0, 0, 0, 0, 0, 0, 0 };
        __builtin_amdgcn_tensor_load_to_lds(g0, g1, gz4, gz4, gz8, 0);
        __builtin_amdgcn_s_wait_tensorcnt(0);
    }
    __syncthreads();
#else
    // ---- Phase A (fallback): 2304 x 16B async chunks, row-padded LDS layout ----
    for (int it = 0; it < 9; ++it) {
        int chunk = it * 256 + tid;                      // 0..2303
        int row   = chunk >> 5;                          // 32 chunks per 128-float row
        int c16   = chunk & 31;
        unsigned lds = (unsigned)(uintptr_t)in_tile + (unsigned)(row * (MWP * 4) + c16 * 16);
        const float* g = src + (size_t)chunk * 4;
        asm volatile("global_load_async_to_lds_b128 %0, %1, off"
                     :: "v"(lds), "v"(g) : "memory");
    }
    asm volatile("s_wait_asynccnt 0" ::: "memory");
    __syncthreads();
#endif

    // ---- Phase B: WMMA width upsample (weights built once per column tile) ----
    const int wave = tid >> 5;
    const int lane = tid & 31;
    const int colL = lane & 15;
    const int hi   = lane >> 4;
    for (int jc = wave; jc < 30; jc += 8) {             // 30 column tiles over 8 waves
        const int c   = jc * 16;
        const int kb  = imin(imax(4 * jc - 1, 0), MW - 32);   // K window base
        const int xu  = c + colL;
        const float inx = (xu + 0.5f) * 0.25f - 0.5f;
        v16h w;
#pragma unroll
        for (int i = 0; i < 16; ++i) {
            // B (32x16 f16) fragment, contiguous k order chosen here (same as A): lanes 0-15 K=0..15, lanes 16-31 K=16..31
            int kg = kb + hi * 16 + i;
            float d  = inx - (float)kg;
            float wt = 1.0f - fabsf(d); wt = wt < 0.0f ? 0.0f : wt;
            if (kg == 0)      { float e = 1.0f - fabsf(inx + 1.0f);      if (e > 0.0f) wt += e; }
            if (kg == MW - 1) { float e = 1.0f - fabsf(inx - (float)MW); if (e > 0.0f) wt += e; }
            w[i] = (_Float16)wt;
        }
        for (int mt_ = 0; mt_ < 5; ++mt_) {             // 80 padded rows / 16
            const int m0   = mt_ * 16;
            const int arow = m0 + colL;                 // rows 72..79 = harmless garbage
            v16h a;
#pragma unroll
            for (int i = 0; i < 16; ++i) {
                // A (16x32 f16) fragment, contiguous k order chosen here (same as B): lanes 0-15 K=0..15, lanes 16-31 K=16..31
                int ka = hi * 16 + i;
                a[i] = (_Float16)in_tile[arow * MWP + kb + ka];
            }
            v8f acc = {};
            acc = __builtin_amdgcn_wmma_f32_16x16x32_f16(false, a, false, w,
                                                         (short)0, acc, false, false);
#pragma unroll
            for (int r = 0; r < 8; ++r)                // unconditional: out_w padded to 80 rows
                out_w[(m0 + hi * 8 + r) * UW + xu] = acc[r];
        }
    }
    __syncthreads();

    // ---- Phase C: height lerp + threshold + 2x2 nearest replicate (b128 NT stores) ----
    float* omask = out + OUT_MASK_OFF + (size_t)slab * ((size_t)OH * OW);
    for (int p = tid; p < UH * (UW / 2); p += 256) {
        int yu = p / (UW / 2), xp = p % (UW / 2);
        int xu = 2 * xp;
        float iny = (yu + 0.5f) * 0.25f - 0.5f;
        float y0f = floorf(iny);
        float fy  = iny - y0f;
        int y0 = imin(imax((int)y0f,     0), MH - 1);
        int y1 = imin(imax((int)y0f + 1, 0), MH - 1);
        float v0 = (1.0f - fy) * out_w[y0 * UW + xu]     + fy * out_w[y1 * UW + xu];
        float v1 = (1.0f - fy) * out_w[y0 * UW + xu + 1] + fy * out_w[y1 * UW + xu + 1];
        float o0 = v0 > 0.0f ? 1.0f : 0.0f;               // sigmoid(v) > 0.5
        float o1 = v1 > 0.0f ? 1.0f : 0.0f;
        f32x4 ov = { o0, o0, o1, o1 };
        float* r0 = omask + (size_t)(2 * yu) * OW + 4 * xp;
        __builtin_nontemporal_store(ov, (f32x4*)r0);
        __builtin_nontemporal_store(ov, (f32x4*)(r0 + OW));
    }
}

// ---------------------------------------------------------------------------
extern "C" void kernel_launch(void* const* d_in, const int* in_sizes, int n_in,
                              void* d_out, int out_size, void* d_ws, size_t ws_size,
                              hipStream_t stream) {
    (void)in_sizes; (void)n_in; (void)out_size; (void)ws_size;
    const float* logits  = (const float*)d_in[0];  // (2,300,42)
    const float* pmasks  = (const float*)d_in[1];  // (2,300,3,72,128)
    const float* ref_pts = (const float*)d_in[2];  // (2,300,4)
    float* out = (float*)d_out;
    int* boxes = (int*)d_ws;                       // 200 ints of scratch

    topk_kernel<<<2, 256, 0, stream>>>(logits, ref_pts, out, boxes);

    const size_t smem = (size_t)(MH * MWP + MHP * UW) * sizeof(float); // 191,616 B
    masks_kernel<<<2 * TOPKN * MT, 256, smem, stream>>>(pmasks, boxes, out);
}
